// DRModel_65472481460952
// MI455X (gfx1250) — matrix operand order repaired and checked by reading, not compiled, not run on hardware
//
#include <hip/hip_runtime.h>
#include <hip/hip_bf16.h>

typedef __attribute__((ext_vector_type(16))) _Float16 v16h;
typedef __attribute__((ext_vector_type(8)))  _Float16 v8h;
typedef __attribute__((ext_vector_type(8)))  float    v8f;

#define B_  256
#define T_  50
#define M_  20
#define E_  128
#define G3  384   // 3*E
#define EP  (E_ + 8)   // padded LDS row: 136 halves = 68 dwords -> bank rotate 4/row

// ---------------- Phase 0: fp32 -> fp16 weight conversion ----------------
__global__ void cvt_f32_f16(const float* __restrict__ src, _Float16* __restrict__ dst, int n) {
  int i = blockIdx.x * blockDim.x + threadIdx.x;
  if (i < n) dst[i] = (_Float16)src[i];
}

// ---------------- Phase 1: embedding gather + mean pool -> ub16 ----------------
// emb[0] == 0 and padded item_ids are 0, so summing all M items is exact.
__global__ void pool_embed(const int* __restrict__ item_ids,
                           const int* __restrict__ basket_sizes,
                           const float* __restrict__ emb,
                           _Float16* __restrict__ ub16) {
  int bt = blockIdx.x;                 // 0..B*T-1
  int e  = threadIdx.x;                // 0..127
  const int* ids = item_ids + (size_t)bt * M_;
  float acc = 0.f;
  #pragma unroll 4
  for (int m = 0; m < M_; ++m) {
    int id = ids[m];
    acc += emb[(size_t)id * E_ + e];
  }
  float bs = (float)basket_sizes[bt];
  ub16[(size_t)bt * E_ + e] = (_Float16)(acc / bs);
}

// ---------------- Phase 2: xg = ub @ W_ih^T + b_ih (WMMA) ----------------
// grid.x = (B*T)/16 M-tiles, grid.y = 3, 8 waves/block -> 24 N-tiles.
__global__ void __launch_bounds__(256)
xg_gemm(const _Float16* __restrict__ A,      // ub16 [B*T][128]
        const _Float16* __restrict__ Wih16,  // [384][128] row-major
        const float* __restrict__ b_ih,      // [384]
        float* __restrict__ xg) {            // [B*T][384]
  const int lane  = threadIdx.x & 31;
  const int wave  = threadIdx.x >> 5;         // 0..7
  const int mtile = blockIdx.x;               // 0..799
  const int ntile = blockIdx.y * 8 + wave;    // 0..23
  const int mrow  = mtile * 16 + (lane & 15);
  const int ncol  = ntile * 16 + (lane & 15);

  v8f acc = {};
  #pragma unroll
  for (int kk = 0; kk < 4; ++kk) {
    // A fragment: lane L<16 -> row M=L, K = kk*32 + 0..15 (contiguous)
    int ka = kk * 32 + ((lane < 16) ? 0 : 16);
    union { v16h v; v8h h[2]; } a;
    const _Float16* ap = A + (size_t)mrow * E_ + ka;
    a.h[0] = *(const v8h*)(ap);
    a.h[1] = *(const v8h*)(ap + 8);
    // B fragment: lane L<16 -> column N=L, K = kk*32 + 0..15 (contiguous)
    int kb = kk * 32 + ((lane < 16) ? 0 : 16);
    v16h b = *(const v16h*)(Wih16 + (size_t)ncol * E_ + kb);
    acc = __builtin_amdgcn_wmma_f32_16x16x32_f16(false, a.v, false, b,
                                                 (short)0, acc, false, false);
  }
  float bias = b_ih[ncol];
  float* out = xg + (size_t)(mtile * 16) * G3 + (size_t)ntile * 16 + (lane & 15);
  #pragma unroll
  for (int r = 0; r < 8; ++r) {
    int mr = r + ((lane < 16) ? 0 : 8);       // C layout: vgpr r -> M=r / M=r+8
    out[(size_t)mr * G3] = acc[r] + bias;
  }
}

// ---------------- Phase 3: persistent batched GRU scan ----------------
// 16 blocks x 256 threads; block handles 16 batch rows for all 50 steps.
// Wave w owns gate columns nc = w*16 + lane%16 across all three gates
// (N-tiles {w, w+8, w+16}), so hr/hz/hn for one element land in the same
// lane's accumulators -> gates fused in WMMA C-layout, h-state in registers,
// single ping-pong LDS buffer + one barrier per step.
__global__ void __launch_bounds__(256, 1)
gru_scan(const _Float16* __restrict__ Whh16,  // [384][128] f16
         const float* __restrict__ xg,        // [B*T][384] (includes b_ih)
         const float* __restrict__ b_hh,      // [384]
         const int* __restrict__ lengths,     // [B]
         const float* __restrict__ h0,        // [B][128]
         float* __restrict__ out_dyn,         // [B][T][128]
         float* __restrict__ out_h) {         // [B][128]
  __shared__ __align__(32) _Float16 h16[2][16][EP];   // ping-pong A staging (f16)

  const int tid   = threadIdx.x;
  const int lane  = tid & 31;
  const int wave  = tid >> 5;               // 0..7
  const int rbase = blockIdx.x * 16;        // batch row base
  const int nc    = wave * 16 + (lane & 15);   // hidden column owned by this lane
  const int hi    = (lane < 16) ? 0 : 8;       // C-layout row offset

  // Resident W_hh B-fragments: gate g -> rows g*128 + nc of W_hh.
  v16h bfrag[3][4];
  #pragma unroll
  for (int g = 0; g < 3; ++g) {
    const _Float16* wrow = Whh16 + (size_t)(g * E_ + nc) * E_;
    #pragma unroll
    for (int kk = 0; kk < 4; ++kk) {
      int kb = kk * 32 + ((lane < 16) ? 0 : 16);
      bfrag[g][kk] = *(const v16h*)(wrow + kb);
    }
  }
  const float bh_r = b_hh[nc];
  const float bh_z = b_hh[E_ + nc];
  const float bh_n = b_hh[2 * E_ + nc];

  // Hidden state + lengths in registers (lane owns rows hi..hi+7, column nc).
  float hstate[8];
  int   len[8];
  #pragma unroll
  for (int r = 0; r < 8; ++r) {
    int mr = r + hi;
    hstate[r] = h0[(size_t)(rbase + mr) * E_ + nc];
    len[r]    = lengths[rbase + mr];
  }
  // Stage initial h into buffer 0.
  for (int idx = tid; idx < 16 * E_; idx += 256) {
    int rr = idx >> 7, e = idx & 127;
    h16[0][rr][e] = (_Float16)h0[(size_t)(rbase + rr) * E_ + e];
  }
  __syncthreads();

  for (int t = 0; t < T_; ++t) {
    const int cur = t & 1, nxt = cur ^ 1;

    // ---- hr/hz/hn = h @ W_hh^T for this lane's column: 12 WMMAs ----
    v8f accr = {}, accz = {}, accn = {};
    #pragma unroll
    for (int kk = 0; kk < 4; ++kk) {
      int ka = kk * 32 + ((lane < 16) ? 0 : 16);
      union { v16h v; v8h h[2]; } a;
      const _Float16* ap = &h16[cur][lane & 15][ka];
      a.h[0] = *(const v8h*)(ap);
      a.h[1] = *(const v8h*)(ap + 8);
      accr = __builtin_amdgcn_wmma_f32_16x16x32_f16(false, a.v, false, bfrag[0][kk],
                                                    (short)0, accr, false, false);
      accz = __builtin_amdgcn_wmma_f32_16x16x32_f16(false, a.v, false, bfrag[1][kk],
                                                    (short)0, accz, false, false);
      accn = __builtin_amdgcn_wmma_f32_16x16x32_f16(false, a.v, false, bfrag[2][kk],
                                                    (short)0, accn, false, false);
    }

    // ---- fused GRU gates directly in WMMA C-layout ----
    #pragma unroll
    for (int r = 0; r < 8; ++r) {
      int mr = r + hi;
      const float* xrow = xg + ((size_t)(rbase + mr) * T_ + t) * G3;
      float xr = xrow[nc], xz = xrow[E_ + nc], xn = xrow[2 * E_ + nc];
      float rg = 1.f / (1.f + __expf(-(xr + accr[r] + bh_r)));
      float zg = 1.f / (1.f + __expf(-(xz + accz[r] + bh_z)));
      float te = __expf(2.f * (xn + rg * (accn[r] + bh_n)));   // tanh via exp
      float ng = (te - 1.f) / (te + 1.f);
      float hnew = (1.f - zg) * ng + zg * hstate[r];
      bool  msk  = (t < len[r]);
      hstate[r] = msk ? hnew : hstate[r];
      h16[nxt][mr][nc] = (_Float16)hstate[r];
      out_dyn[((size_t)(rbase + mr) * T_ + t) * E_ + nc] = msk ? hnew : 0.f;
    }
    __syncthreads();   // next step reads h16[nxt]; writes never touch h16[cur]
  }

  // final hidden state straight from registers
  #pragma unroll
  for (int r = 0; r < 8; ++r) {
    int mr = r + hi;
    out_h[(size_t)(rbase + mr) * E_ + nc] = hstate[r];
  }
}

extern "C" void kernel_launch(void* const* d_in, const int* in_sizes, int n_in,
                              void* d_out, int out_size, void* d_ws, size_t ws_size,
                              hipStream_t stream) {
  const int*   item_ids     = (const int*)d_in[0];
  const int*   basket_sizes = (const int*)d_in[1];
  const int*   lengths      = (const int*)d_in[2];
  const float* emb          = (const float*)d_in[3];
  const float* W_ih         = (const float*)d_in[4];
  const float* W_hh         = (const float*)d_in[5];
  const float* b_ih         = (const float*)d_in[6];
  const float* b_hh         = (const float*)d_in[7];
  const float* h0           = (const float*)d_in[8];
  (void)in_sizes; (void)n_in; (void)out_size; (void)ws_size;

  char* ws = (char*)d_ws;
  size_t off = 0;
  auto alloc = [&](size_t bytes) -> void* {
    void* p = ws + off;
    off = (off + bytes + 255) & ~(size_t)255;
    return p;
  };
  _Float16* Wih16 = (_Float16*)alloc((size_t)G3 * E_ * sizeof(_Float16));
  _Float16* Whh16 = (_Float16*)alloc((size_t)G3 * E_ * sizeof(_Float16));
  _Float16* ub16  = (_Float16*)alloc((size_t)B_ * T_ * E_ * sizeof(_Float16));
  float*    xg    = (float*)   alloc((size_t)B_ * T_ * G3 * sizeof(float));

  cvt_f32_f16<<<(G3 * E_ + 255) / 256, 256, 0, stream>>>(W_ih, Wih16, G3 * E_);
  cvt_f32_f16<<<(G3 * E_ + 255) / 256, 256, 0, stream>>>(W_hh, Whh16, G3 * E_);

  pool_embed<<<B_ * T_, E_, 0, stream>>>(item_ids, basket_sizes, emb, ub16);

  dim3 g2(B_ * T_ / 16, 3);
  xg_gemm<<<g2, 256, 0, stream>>>(ub16, Wih16, b_ih, xg);

  float* out_dyn = (float*)d_out;
  float* out_h   = out_dyn + (size_t)B_ * T_ * E_;
  gru_scan<<<B_ / 16, 256, 0, stream>>>(Whh16, xg, b_hh, lengths, h0, out_dyn, out_h);
}
